// RelativeMultiHeadAttention_74113955660145
// MI455X (gfx1250) — matrix operand order repaired and checked by reading, not compiled, not run on hardware
//
#include <hip/hip_runtime.h>
#include <hip/hip_bf16.h>
#include <math.h>

#define S_LEN 2048
#define DM    1024
#define NH    16
#define DH    64

typedef __attribute__((ext_vector_type(16))) __bf16 bf16x16;
typedef __attribute__((ext_vector_type(8)))  __bf16 bf16x8;
typedef __attribute__((ext_vector_type(8)))  float  f32x8;

__device__ __forceinline__ f32x8 wmma_bf16(bf16x16 a, bf16x16 b, f32x8 c) {
    return __builtin_amdgcn_wmma_f32_16x16x32_bf16(false, a, false, b, (short)0, c,
                                                   false, false);
}

// Concatenate two contiguous 16-byte runs into one fragment register set.
__device__ __forceinline__ bf16x16 frag_cat(const __bf16* p0, const __bf16* p1) {
    bf16x8 a = *(const bf16x8*)p0;   // global_load_b128
    bf16x8 b = *(const bf16x8*)p1;   // global_load_b128
    return __builtin_shufflevector(a, b, 0, 1, 2, 3, 4, 5, 6, 7,
                                         8, 9, 10, 11, 12, 13, 14, 15);
}

// A-fragment (16x32 bf16, row-major source): lane half h needs K runs
// [k0+8h, k0+8h+8) and [k0+16+8h, k0+16+8h+8)  (ISA 7.12.2 16-bit A layout)
__device__ __forceinline__ bf16x16 load_a_frag(const __bf16* row, int k0, int half) {
    const __bf16* p = row + k0 + 8 * half;
    return frag_cat(p, p + 16);
}

// B-fragment (32x16 bf16, K-major source with row length ldb): lane holds column n0+lane%16,
// elements gather K in table order (8h+s, 16+8h+s-8) matching the A fragment.
__device__ __forceinline__ bf16x16 load_b_frag(const __bf16* base, int k, int ldb, int n0) {
    int h = (threadIdx.x >> 4) & 1;
    const __bf16* p = base + (size_t)k * ldb + n0 + (threadIdx.x & 15);
    bf16x16 r;
#pragma unroll
    for (int s = 0; s < 16; ++s) {
        int kk = (s < 8) ? (8 * h + s) : (16 + 8 * h + (s - 8));
        r[s] = p[(size_t)kk * ldb];
    }
    return r;
}

// A-fragment from f32 source with on-the-fly bf16 convert (for weights @ V).
__device__ __forceinline__ bf16x16 load_a_frag_f32(const float* row, int k0, int half) {
    const float4* p0 = (const float4*)(row + k0 + 8 * half);
    const float4* p1 = (const float4*)(row + k0 + 16 + 8 * half);
    float4 f0 = p0[0], f1 = p0[1], f2 = p1[0], f3 = p1[1];
    bf16x16 r;
    r[0] = (__bf16)f0.x;  r[1] = (__bf16)f0.y;  r[2] = (__bf16)f0.z;  r[3] = (__bf16)f0.w;
    r[4] = (__bf16)f1.x;  r[5] = (__bf16)f1.y;  r[6] = (__bf16)f1.z;  r[7] = (__bf16)f1.w;
    r[8] = (__bf16)f2.x;  r[9] = (__bf16)f2.y;  r[10] = (__bf16)f2.z; r[11] = (__bf16)f2.w;
    r[12] = (__bf16)f3.x; r[13] = (__bf16)f3.y; r[14] = (__bf16)f3.z; r[15] = (__bf16)f3.w;
    return r;
}

// ---------------- format conversion kernels ----------------

// Wt[k][n] = (bf16) W[n][k]
__global__ void wt_convert_kernel(const float* __restrict__ W, __bf16* __restrict__ Wt) {
    int idx = blockIdx.x * blockDim.x + threadIdx.x;
    if (idx >= DM * DM) return;
    int k = idx / DM, n = idx % DM;
    Wt[idx] = (__bf16)W[n * DM + k];
}

__global__ void f32_to_bf16_kernel(const float* __restrict__ X, __bf16* __restrict__ Xb, int n) {
    int i = blockIdx.x * blockDim.x + threadIdx.x;
    if (i < n) Xb[i] = (__bf16)X[i];
}

// reversed sinusoidal positional encoding -> bf16
__global__ void rel_kernel(__bf16* __restrict__ rel) {
    int i = blockIdx.x * blockDim.x + threadIdx.x;
    if (i >= S_LEN * DM) return;
    int f = i / DM, c = i % DM;
    float pos = (float)(S_LEN - 1 - f);
    float e2  = (float)(c & ~1);
    float ang = pos * expf(-(e2 / (float)DM) * 9.210340371976184f); // ln(10000)
    rel[i] = (__bf16)((c & 1) ? cosf(ang) : sinf(ang));
}

// qu = qproj + u_param (per column), qv = qproj + v_param, both to bf16
__global__ void quqv_kernel(const float* __restrict__ qproj, const float* __restrict__ u,
                            const float* __restrict__ vp, __bf16* __restrict__ qu,
                            __bf16* __restrict__ qv) {
    int i = blockIdx.x * blockDim.x + threadIdx.x;
    if (i >= S_LEN * DM) return;
    int c = i & (DM - 1);
    float q = qproj[i];
    qu[i] = (__bf16)(q + u[c]);
    qv[i] = (__bf16)(q + vp[c]);
}

// XT[h][k][f] = X[f][h*64+k]   ((S,DM) -> (NH,DH,S)) so attention B-loads are contiguous
__global__ void headT_kernel(const __bf16* __restrict__ X, __bf16* __restrict__ XT) {
    int i = blockIdx.x * blockDim.x + threadIdx.x;
    if (i >= S_LEN * DM) return;
    int f  = i % S_LEN;
    int hk = i / S_LEN;     // = h*DH + k
    XT[i] = X[(size_t)f * DM + hk];
}

// ---------------- WMMA GEMM kernels ----------------

// C[M,N] = A[M,K](bf16) @ Bt[K,N](bf16); 16x64 tile per wave (A reused 4x,
// 4 independent accumulators hide the wmma->wmma RAW hazard).
// MODE: 0=f32 out, 1=bf16 out, 2=f32+exact-GELU
template <int MODE>
__global__ __launch_bounds__(32) void gemm_bt_kernel(const __bf16* __restrict__ A,
        const __bf16* __restrict__ Bt, float* __restrict__ outF, __bf16* __restrict__ outB,
        int M, int N, int K, int lda, int ldb, int ldc) {
    int lane = threadIdx.x, half = lane >> 4;
    int m0 = blockIdx.y * 16, n0 = blockIdx.x * 64;
    const __bf16* arow = A + (size_t)(m0 + (lane & 15)) * lda;
    f32x8 acc[4] = {};
    for (int k0 = 0; k0 < K; k0 += 32) {
        if (k0 + 32 < K) __builtin_prefetch(arow + k0 + 32, 0, 0);  // global_prefetch_b8
        bf16x16 a = load_a_frag(arow, k0, half);
#pragma unroll
        for (int j = 0; j < 4; ++j) {
            bf16x16 b = load_b_frag(Bt, k0, ldb, n0 + 16 * j);
            acc[j] = wmma_bf16(a, b, acc[j]);
        }
    }
#pragma unroll
    for (int j = 0; j < 4; ++j) {
        int n = n0 + 16 * j + (lane & 15);
#pragma unroll
        for (int r = 0; r < 8; ++r) {
            int m = m0 + r + 8 * half;
            float v = acc[j][r];
            if (MODE == 2) v = 0.5f * v * (1.0f + erff(v * 0.7071067811865475f));
            if (MODE == 1) outB[(size_t)m * ldc + n] = (__bf16)v;
            else           outF[(size_t)m * ldc + n] = v;
        }
    }
}

// B_hat[h][s][f] = (q+v)[s, h*64:] . qrelT[h][:,f]   (qrelT is (NH,DH,S))
__global__ __launch_bounds__(32) void bhat_kernel(const __bf16* __restrict__ qv,
        const __bf16* __restrict__ qrelT, float* __restrict__ bhat) {
    int lane = threadIdx.x, half = lane >> 4;
    int f0 = blockIdx.x * 16, s0 = blockIdx.y * 16, h = blockIdx.z;
    const __bf16* arow = qv + (size_t)(s0 + (lane & 15)) * DM + h * DH;
    const __bf16* BbT  = qrelT + (size_t)h * DH * S_LEN;
    f32x8 acc = {};
#pragma unroll
    for (int k0 = 0; k0 < DH; k0 += 32) {
        bf16x16 a = load_a_frag(arow, k0, half);
        bf16x16 b = load_b_frag(BbT, k0, S_LEN, f0);
        acc = wmma_bf16(a, b, acc);
    }
    float* out = bhat + (size_t)h * S_LEN * S_LEN;
    int f = f0 + (lane & 15);
#pragma unroll
    for (int r = 0; r < 8; ++r) {
        int s = s0 + r + 8 * half;
        out[(size_t)s * S_LEN + f] = acc[r];
    }
}

// score = (AC + rel_shift(B_hat)) / 32 + mask -> weights tensor (pre-softmax)
__global__ __launch_bounds__(32) void score_kernel(const __bf16* __restrict__ qu,
        const __bf16* __restrict__ kprojT, const float* __restrict__ bhat,
        const float* __restrict__ mask, float* __restrict__ weights) {
    int lane = threadIdx.x, half = lane >> 4;
    int f0 = blockIdx.x * 16, s0 = blockIdx.y * 16, h = blockIdx.z;
    const __bf16* arow = qu + (size_t)(s0 + (lane & 15)) * DM + h * DH;
    const __bf16* BbT  = kprojT + (size_t)h * DH * S_LEN;
    f32x8 acc = {};
#pragma unroll
    for (int k0 = 0; k0 < DH; k0 += 32) {
        bf16x16 a = load_a_frag(arow, k0, half);
        bf16x16 b = load_b_frag(BbT, k0, S_LEN, f0);
        acc = wmma_bf16(a, b, acc);
    }
    const float* bh = bhat + (size_t)h * S_LEN * S_LEN;
    float* wout = weights + (size_t)h * S_LEN * S_LEN;
    int f = f0 + (lane & 15);
#pragma unroll
    for (int r = 0; r < 8; ++r) {
        int s = s0 + r + 8 * half;
        // exact rel_enc_shift gather: flat t = s*S + f + (S-1) over an (S, S+1) view
        long long t = (long long)s * S_LEN + f + (S_LEN - 1);
        int srow = (int)(t / (S_LEN + 1));
        int scol = (int)(t % (S_LEN + 1));
        float bd = (scol < S_LEN) ? bh[(size_t)srow * S_LEN + scol] : 0.0f;
        float sc = (acc[r] + bd) * 0.03125f + mask[(size_t)s * S_LEN + f];
        wout[(size_t)s * S_LEN + f] = sc;
    }
}

// out_pre[s, h*64:h*64+64] = weights[h,s,:] @ vproj[:, h*64:h*64+64]
// One wave per (s-tile, head): weights row panel is read exactly ONCE (4 N-tiles
// share the converted A fragment; 4 independent accumulators).
__global__ __launch_bounds__(32) void av_kernel(const float* __restrict__ weights,
        const __bf16* __restrict__ vproj, float* __restrict__ outpre) {
    int lane = threadIdx.x, half = lane >> 4;
    int s0 = blockIdx.x * 16, h = blockIdx.y;
    int n0 = h * DH;
    const float* arow = weights + (size_t)h * S_LEN * S_LEN
                                + (size_t)(s0 + (lane & 15)) * S_LEN;
    f32x8 acc[4] = {};
    for (int k0 = 0; k0 < S_LEN; k0 += 32) {
        if (k0 + 32 < S_LEN) __builtin_prefetch(arow + k0 + 32, 0, 0);
        bf16x16 a = load_a_frag_f32(arow, k0, half);
#pragma unroll
        for (int j = 0; j < 4; ++j) {
            bf16x16 b = load_b_frag(vproj, k0, DM, n0 + 16 * j);
            acc[j] = wmma_bf16(a, b, acc[j]);
        }
    }
#pragma unroll
    for (int j = 0; j < 4; ++j) {
        int n = n0 + 16 * j + (lane & 15);
#pragma unroll
        for (int r = 0; r < 8; ++r) {
            int m = s0 + r + 8 * half;
            outpre[(size_t)m * DM + n] = acc[j][r];
        }
    }
}

// ---------------- softmax / loss ----------------

__global__ __launch_bounds__(256) void softmax_kernel(float* __restrict__ weights,
                                                      float* __restrict__ rowinv) {
    __shared__ float sdata[256];
    int row = blockIdx.x;                 // row in [0, NH*S)
    float* w = weights + (size_t)row * S_LEN;
    int tid = threadIdx.x;
    float loc[8];
    float m = -3.4e38f;
#pragma unroll
    for (int i = 0; i < 8; ++i) { loc[i] = w[tid + 256 * i]; m = fmaxf(m, loc[i]); }
    sdata[tid] = m; __syncthreads();
    for (int st = 128; st > 0; st >>= 1) {
        if (tid < st) sdata[tid] = fmaxf(sdata[tid], sdata[tid + st]);
        __syncthreads();
    }
    m = sdata[0]; __syncthreads();
    float z = 0.0f;
#pragma unroll
    for (int i = 0; i < 8; ++i) { loc[i] = expf(loc[i] - m); z += loc[i]; }
    sdata[tid] = z; __syncthreads();
    for (int st = 128; st > 0; st >>= 1) {
        if (tid < st) sdata[tid] += sdata[tid + st];
        __syncthreads();
    }
    float inv = 1.0f / sdata[0];
#pragma unroll
    for (int i = 0; i < 8; ++i) w[tid + 256 * i] = loc[i] * inv;
    if (tid == 0) rowinv[row] = inv;       // max of a softmax row == 1/Z
}

__global__ __launch_bounds__(256) void loss_kernel(const float* __restrict__ rowinv,
                                                   float* __restrict__ lossOut) {
    __shared__ float sdata[256];
    int tid = threadIdx.x;
    float acc = 0.0f;
    for (int h = 0; h < NH; ++h) {
        float m = 0.0f;
        for (int s = tid; s < S_LEN; s += 256) m = fmaxf(m, rowinv[h * S_LEN + s]);
        sdata[tid] = m; __syncthreads();
        for (int st = 128; st > 0; st >>= 1) {
            if (tid < st) sdata[tid] = fmaxf(sdata[tid], sdata[tid + st]);
            __syncthreads();
        }
        if (tid == 0) acc += sdata[0];
        __syncthreads();
    }
    if (tid == 0) *lossOut = acc / (float)NH;
}

// ---------------- host launcher ----------------

extern "C" void kernel_launch(void* const* d_in, const int* in_sizes, int n_in,
                              void* d_out, int out_size, void* d_ws, size_t ws_size,
                              hipStream_t stream) {
    const float* query = (const float*)d_in[0];
    const float* key   = (const float*)d_in[1];
    const float* value = (const float*)d_in[2];
    const float* mask  = (const float*)d_in[3];
    const float* Wq    = (const float*)d_in[4];
    const float* Wke   = (const float*)d_in[5];
    const float* Wkr   = (const float*)d_in[6];
    const float* Wv    = (const float*)d_in[7];
    const float* Wf    = (const float*)d_in[8];
    const float* up    = (const float*)d_in[9];
    const float* vp    = (const float*)d_in[10];

    float* outp    = (float*)d_out;                       // (S, D) f32
    float* weights = outp + (size_t)S_LEN * DM;           // (H, S, S) f32
    float* lossp   = weights + (size_t)NH * S_LEN * S_LEN;

    char* ws = (char*)d_ws;
    size_t off = 0;
    auto take = [&](size_t bytes) -> char* {
        char* p = ws + off;
        off += (bytes + 255) & ~(size_t)255;
        return p;
    };
    __bf16* wq_t   = (__bf16*)take((size_t)DM * DM * 2);
    __bf16* wke_t  = (__bf16*)take((size_t)DM * DM * 2);
    __bf16* wkr_t  = (__bf16*)take((size_t)DM * DM * 2);
    __bf16* wv_t   = (__bf16*)take((size_t)DM * DM * 2);
    __bf16* wf_t   = (__bf16*)take((size_t)DM * DM * 2);
    __bf16* q_bf   = (__bf16*)take((size_t)S_LEN * DM * 2);   // reused for out_pre bf16
    __bf16* k_bf   = (__bf16*)take((size_t)S_LEN * DM * 2);
    __bf16* v_bf   = (__bf16*)take((size_t)S_LEN * DM * 2);
    __bf16* rel_bf = (__bf16*)take((size_t)S_LEN * DM * 2);
    __bf16* qrel   = (__bf16*)take((size_t)S_LEN * DM * 2);
    __bf16* kproj  = (__bf16*)take((size_t)S_LEN * DM * 2);
    __bf16* vproj  = (__bf16*)take((size_t)S_LEN * DM * 2);
    __bf16* qu     = (__bf16*)take((size_t)S_LEN * DM * 2);
    __bf16* qv     = (__bf16*)take((size_t)S_LEN * DM * 2);
    __bf16* kprojT = (__bf16*)take((size_t)S_LEN * DM * 2);   // (NH, DH, S)
    __bf16* qrelT  = (__bf16*)take((size_t)S_LEN * DM * 2);   // (NH, DH, S)
    float*  qproj  = (float*) take((size_t)S_LEN * DM * 4);
    float*  bhat   = (float*) take((size_t)NH * S_LEN * S_LEN * 4);
    float*  outpre = (float*) take((size_t)S_LEN * DM * 4);
    float*  rowinv = (float*) take((size_t)NH * S_LEN * 4);

    const int SD = S_LEN * DM, DD = DM * DM;

    // 1) convert weights (transposed) and activations to bf16; build rel encoding
    wt_convert_kernel<<<DD / 256, 256, 0, stream>>>(Wq,  wq_t);
    wt_convert_kernel<<<DD / 256, 256, 0, stream>>>(Wke, wke_t);
    wt_convert_kernel<<<DD / 256, 256, 0, stream>>>(Wkr, wkr_t);
    wt_convert_kernel<<<DD / 256, 256, 0, stream>>>(Wv,  wv_t);
    wt_convert_kernel<<<DD / 256, 256, 0, stream>>>(Wf,  wf_t);
    f32_to_bf16_kernel<<<SD / 256, 256, 0, stream>>>(query, q_bf, SD);
    f32_to_bf16_kernel<<<SD / 256, 256, 0, stream>>>(key,   k_bf, SD);
    f32_to_bf16_kernel<<<SD / 256, 256, 0, stream>>>(value, v_bf, SD);
    rel_kernel<<<SD / 256, 256, 0, stream>>>(rel_bf);

    // 2) projections (WMMA, 16x64 tile per wave)
    dim3 gP(DM / 64, S_LEN / 16);
    gemm_bt_kernel<0><<<gP, 32, 0, stream>>>(q_bf,   wq_t,  qproj,   nullptr, S_LEN, DM, DM, DM, DM, DM);
    gemm_bt_kernel<1><<<gP, 32, 0, stream>>>(k_bf,   wke_t, nullptr, kproj,   S_LEN, DM, DM, DM, DM, DM);
    gemm_bt_kernel<1><<<gP, 32, 0, stream>>>(v_bf,   wv_t,  nullptr, vproj,   S_LEN, DM, DM, DM, DM, DM);
    gemm_bt_kernel<1><<<gP, 32, 0, stream>>>(rel_bf, wkr_t, nullptr, qrel,    S_LEN, DM, DM, DM, DM, DM);
    quqv_kernel<<<SD / 256, 256, 0, stream>>>(qproj, up, vp, qu, qv);

    // per-head transposes so attention B-fragment element loads are contiguous across lanes
    headT_kernel<<<SD / 256, 256, 0, stream>>>(kproj, kprojT);
    headT_kernel<<<SD / 256, 256, 0, stream>>>(qrel,  qrelT);

    // 3) per-head position scores + fused content-score/shift/mask (WMMA)
    dim3 gS(S_LEN / 16, S_LEN / 16, NH);
    bhat_kernel <<<gS, 32, 0, stream>>>(qv, qrelT, bhat);
    score_kernel<<<gS, 32, 0, stream>>>(qu, kprojT, bhat, mask, weights);

    // 4) softmax in place on the weights output; row maxima (=1/Z) -> loss
    softmax_kernel<<<NH * S_LEN, 256, 0, stream>>>(weights, rowinv);
    loss_kernel<<<1, 256, 0, stream>>>(rowinv, lossp);

    // 5) attention-weighted values (WMMA, one pass over weights), then
    //    final projection + exact GELU (WMMA)
    dim3 gAV(S_LEN / 16, NH);
    av_kernel<<<gAV, 32, 0, stream>>>(weights, vproj, outpre);
    f32_to_bf16_kernel<<<SD / 256, 256, 0, stream>>>(outpre, q_bf, SD);  // reuse q_bf
    gemm_bt_kernel<2><<<gP, 32, 0, stream>>>(q_bf, wf_t, outp, nullptr, S_LEN, DM, DM, DM, DM, DM);

    (void)in_sizes; (void)n_in; (void)out_size; (void)ws_size;
}
